// TensorBase_71279277245010
// MI455X (gfx1250) — matrix operand order repaired and checked by reading, not compiled, not run on hardware
//
#include <hip/hip_runtime.h>
#include <hip/hip_bf16.h>

// ---------------------------------------------------------------------------
// TensoRF render for MI455X (gfx1250): featurize (gather + PE) -> WMMA f16 MLP
// MLP weights staged once per block into LDS via async global->LDS loads so
// WMMA B operands come from ds_load instead of 8x-replicated L2 streams.
// K-tile loops kept rolled (#pragma unroll 1) to stay under 256 VGPRs:
// avoids s_set_vgpr_msb traffic and scratch spills seen with full unroll.
// Assumes ws_size >= 28 MB (A-fragment chunk 27.3MB + weights + sigma).
// ---------------------------------------------------------------------------

typedef __attribute__((ext_vector_type(16))) _Float16 v16h;
typedef __attribute__((ext_vector_type(8)))  float    v8f;
typedef __attribute__((ext_vector_type(4)))  int      v4i;

#define NR       4096
#define NS       128
#define GS       128
#define VOL      (GS*GS*GS)
#define APP_DIM  27
#define FC       128
#define IN_DIM   390
#define KT1      13            // ceil(390/32) K-tiles for layer 1 (K padded to 416)
#define NT1      8             // 128/16 N-tiles
#define KT2      4             // 128/32 K-tiles for layers 2,3
#define CH_RAYS  256           // rays per chunk
#define CH_PTS   (CH_RAYS*NS)  // 32768 points per chunk
#define NCHUNK   (NR/CH_RAYS)  // 16

// ---- packed weight sizes (halves) ----
#define W1F_H    (KT1*NT1*512)           // 53248
#define W2F_H    (KT2*NT1*512)           // 16384
#define W3F_H    (KT2*512)               //  2048
#define WALL_H   (W1F_H+W2F_H+W3F_H)     // 71680 halves = 143360 B

// ---- workspace layout (bytes) ----
#define WS_WALL  0                       // packed w1f|w2f|w3f  = 143360 B
#define WS_SIG   143360                  // 32768 floats        = 131072 B
#define WS_AFRAG 274432                  // 2048*13*512 halves  = 27262976 B

#define AS1 __attribute__((address_space(1)))
#define AS3 __attribute__((address_space(3)))

#if defined(__has_builtin)
#if __has_builtin(__builtin_amdgcn_global_load_async_to_lds_b128)
#define HAVE_ASYNC_LDS 1
#endif
#endif

// ---------------------------------------------------------------------------
// Fragment index helpers, per CDNA5 ISA 7.12.2 (wave32, 16-bit operands).
// A (16x32, row m, k): lane = m + 16*((k>>3)&1); j = (k&7) | ((k&16)>>1)
// B (32x16, k, col n): lane = n + 16*((k>>3)&1); j = (k&7) | ((k&16)>>1)
// Fragment stored as frag[lane*16 + j] -> each lane loads 16 contiguous halves.
// ---------------------------------------------------------------------------
__device__ __host__ __forceinline__ int afrag_idx(int tile, int m, int k) {
  int kt   = k >> 5;
  int kl   = k & 31;
  int lane = m + ((kl & 8) ? 16 : 0);
  int j    = (kl & 7) | ((kl & 16) >> 1);
  return (tile * KT1 + kt) * 512 + lane * 16 + j;
}

// ---------------------------------------------------------------------------
// Kernel 1: repack MLP weights f32 -> fp16 WMMA B-fragments (run once).
// Output layout: [w1f | w2f | w3f] contiguous so the GEMM kernel can stage it
// into LDS with one async bulk copy.
// ---------------------------------------------------------------------------
__global__ void prep_weights(const float* __restrict__ w1,
                             const float* __restrict__ w2,
                             const float* __restrict__ w3,
                             _Float16* __restrict__ wall) {
  int i = blockIdx.x * blockDim.x + threadIdx.x;
  if (i < W1F_H) {
    int frag = i >> 9, pos = i & 511;
    int lane = pos >> 4, j = pos & 15;
    int kt = frag / NT1, nt = frag % NT1;
    int k = kt * 32 + ((lane >= 16) ? 8 : 0) + (j & 7) + ((j & 8) << 1);
    int n = nt * 16 + (lane & 15);
    wall[i] = (_Float16)((k < IN_DIM) ? w1[k * FC + n] : 0.0f);
  } else if (i < W1F_H + W2F_H) {
    int e = i - W1F_H;
    int frag = e >> 9, pos = e & 511;
    int lane = pos >> 4, j = pos & 15;
    int kt = frag / NT1, nt = frag % NT1;
    int k = kt * 32 + ((lane >= 16) ? 8 : 0) + (j & 7) + ((j & 8) << 1);
    int n = nt * 16 + (lane & 15);
    wall[i] = (_Float16)w2[k * FC + n];
  } else if (i < WALL_H) {
    int e = i - W1F_H - W2F_H;
    int kt = e >> 9, pos = e & 511;
    int lane = pos >> 4, j = pos & 15;
    int k = kt * 32 + ((lane >= 16) ? 8 : 0) + (j & 7) + ((j & 8) << 1);
    int n = lane & 15;
    wall[i] = (_Float16)((n < 3) ? w3[k * 3 + n] : 0.0f);
  }
}

// ---------------------------------------------------------------------------
// Kernel 2: per-point featurization for one chunk.
// Trilinear density -> softplus sigma; 27-ch trilinear app features; positional
// encodings. Features written straight into WMMA A-fragment layout (fp16).
// ---------------------------------------------------------------------------
__global__ void __launch_bounds__(256)
featurize(const float* __restrict__ xyz,
          const float* __restrict__ viewdirs,
          const float* __restrict__ density_grid,
          const float* __restrict__ app_grid,
          _Float16* __restrict__ afrag,
          float* __restrict__ sigma,
          int ray0) {
  int pc = blockIdx.x * blockDim.x + threadIdx.x;   // point within chunk
  if (pc >= CH_PTS) return;
  int p   = ray0 * NS + pc;                          // global point
  int ray = p / NS;

  float x = xyz[p * 3 + 0], y = xyz[p * 3 + 1], z = xyz[p * 3 + 2];

  // align_corners trilinear setup (x->W, y->H, z->D)
  float fx = (x + 1.0f) * 0.5f * (GS - 1);
  float fy = (y + 1.0f) * 0.5f * (GS - 1);
  float fz = (z + 1.0f) * 0.5f * (GS - 1);
  int x0 = (int)floorf(fx); x0 = x0 < 0 ? 0 : (x0 > GS - 2 ? GS - 2 : x0);
  int y0 = (int)floorf(fy); y0 = y0 < 0 ? 0 : (y0 > GS - 2 ? GS - 2 : y0);
  int z0 = (int)floorf(fz); z0 = z0 < 0 ? 0 : (z0 > GS - 2 ? GS - 2 : z0);
  float wx = fx - (float)x0, wy = fy - (float)y0, wz = fz - (float)z0;

  float cw[8]; int coff[8];
  #pragma unroll
  for (int i = 0; i < 8; ++i) {
    int dx = i & 1, dy = (i >> 1) & 1, dz = (i >> 2) & 1;
    cw[i]   = (dx ? wx : 1.0f - wx) * (dy ? wy : 1.0f - wy) * (dz ? wz : 1.0f - wz);
    coff[i] = ((z0 + dz) * GS + (y0 + dy)) * GS + (x0 + dx);
  }

  // density -> softplus(sigma_feat - 10)
  float sf = 0.0f;
  #pragma unroll
  for (int i = 0; i < 8; ++i) sf += cw[i] * density_grid[coff[i]];
  float sp_in = sf - 10.0f;
  sigma[pc] = (sp_in > 20.0f) ? sp_in : log1pf(__expf(sp_in));

  // 27-channel appearance features
  float ft[APP_DIM];
  #pragma unroll
  for (int c = 0; c < APP_DIM; ++c) {
    const float* g = app_grid + (size_t)c * VOL;
    float v = 0.0f;
    #pragma unroll
    for (int i = 0; i < 8; ++i) v += cw[i] * g[coff[i]];
    ft[c] = v;
  }
  float vd[3];
  #pragma unroll
  for (int d = 0; d < 3; ++d) vd[d] = viewdirs[ray * 3 + d];

  int tile = pc >> 4, m = pc & 15;
  auto wf = [&](int k, float v) {
    afrag[afrag_idx(tile, m, k)] = (_Float16)v;
  };

  // mlp_in layout: [feats 27 | vdirs 3 | sin(PE fea) 162 | cos 162 | sin(PE vd) 18 | cos 18]
  #pragma unroll
  for (int c = 0; c < APP_DIM; ++c) wf(c, ft[c]);
  #pragma unroll
  for (int d = 0; d < 3; ++d) wf(27 + d, vd[d]);
  #pragma unroll
  for (int c = 0; c < APP_DIM; ++c) {
    #pragma unroll
    for (int f = 0; f < 6; ++f) {
      float a = ft[c] * (float)(1 << f);
      wf(30 + c * 6 + f, __sinf(a));
      wf(192 + c * 6 + f, __cosf(a));
    }
  }
  #pragma unroll
  for (int d = 0; d < 3; ++d) {
    #pragma unroll
    for (int f = 0; f < 6; ++f) {
      float a = vd[d] * (float)(1 << f);
      wf(354 + d * 6 + f, __sinf(a));
      wf(372 + d * 6 + f, __cosf(a));
    }
  }
  #pragma unroll
  for (int k = IN_DIM; k < KT1 * 32; ++k) wf(k, 0.0f);  // K padding
}

// ---------------------------------------------------------------------------
// Kernel 3: one block per ray (8 waves x 16-sample tile). WMMA MLP + composite.
// Weights async-staged to LDS once per block; B operands served by ds_load.
// ---------------------------------------------------------------------------
__global__ void __launch_bounds__(256)
mlp_render(const _Float16* __restrict__ afrag,
           const _Float16* __restrict__ wall,   // packed w1f|w2f|w3f fragments
           const float* __restrict__ b1,
           const float* __restrict__ b2,
           const float* __restrict__ b3,
           const float* __restrict__ sigma,
           const float* __restrict__ dists,
           float* __restrict__ out,
           int ray0) {
  __shared__ _Float16 wlds[WALL_H];        // 143360 B packed weight fragments
  __shared__ _Float16 hbuf[8 * 16 * FC];   // per-wave 16x128 fp16 activations (32 KB)
  __shared__ float    wray[NS];            // compositing weights for this ray
  __shared__ float    wsum[8 * 3];         // per-wave rgb partials

  const int tid  = threadIdx.x;
  const int wave = tid >> 5;
  const int lane = tid & 31;
  const int rayc = blockIdx.x;             // ray within chunk
  const int ray  = ray0 + rayc;

  // ---- stage packed weights to LDS (async global->LDS, 16B per lane) ----
  {
    const v4i* gsrc = (const v4i*)wall;
    v4i*       ldst = (v4i*)wlds;
    #pragma unroll 1
    for (int i = tid; i < WALL_H / 8; i += 256) {
#if defined(HAVE_ASYNC_LDS)
      __builtin_amdgcn_global_load_async_to_lds_b128(
          (AS1 v4i*)(gsrc + i), (AS3 v4i*)(ldst + i), 0, 0);
#else
      ldst[i] = gsrc[i];
#endif
    }
  }

  // ---- wave 0: parallel transmittance scan (4 samples/lane + shfl scan) ----
  if (wave == 0) {
    float om[4], wpre[4];
    float lp = 1.0f;
    #pragma unroll
    for (int i = 0; i < 4; ++i) {
      int s = lane * 4 + i;
      float sg    = sigma[rayc * NS + s];
      float alpha = 1.0f - __expf(-sg * dists[ray * NS + s] * 25.0f);
      wpre[i] = alpha;
      om[i]   = 1.0f - alpha + 1e-10f;
      lp *= om[i];
    }
    float scan = lp;                        // inclusive prefix product over lanes
    #pragma unroll
    for (int d = 1; d < 32; d <<= 1) {
      float t = __shfl_up(scan, d);
      if (lane >= d) scan *= t;
    }
    float T = __shfl_up(scan, 1);           // exclusive prefix
    if (lane == 0) T = 1.0f;
    #pragma unroll
    for (int i = 0; i < 4; ++i) {
      wray[lane * 4 + i] = wpre[i] * T;
      T *= om[i];
    }
  }

#if defined(HAVE_ASYNC_LDS)
#if __has_builtin(__builtin_amdgcn_s_wait_asynccnt)
  __builtin_amdgcn_s_wait_asynccnt(0);
#else
  asm volatile("s_wait_asynccnt 0" ::: "memory");
#endif
#endif
  __syncthreads();

  const _Float16* w1l = wlds;
  const _Float16* w2l = wlds + W1F_H;
  const _Float16* w3l = wlds + W1F_H + W2F_H;

  const int tilec = rayc * 8 + wave;       // 16-row tile within chunk
  const _Float16* aBase = afrag + (size_t)tilec * KT1 * 512 + lane * 16;

  const v8f zero = {};
  const int nlo = lane & 15;               // C/D column within 16-tile
  const int mhi = (lane >> 4) << 3;        // C/D row offset (0 or 8)
  const int khi = (lane & 16) >> 1;        // A-frag K offset (0 or 8)
  const int mA  = lane & 15;               // A-frag row for this lane

  // ---------------- layer 1: [16x416] x [416x128] ----------------
  v8f acc[NT1];
  #pragma unroll
  for (int nt = 0; nt < NT1; ++nt) acc[nt] = zero;
  #pragma unroll 1                         // rolled: bounds live B-frags < 256 VGPRs
  for (int kt = 0; kt < KT1; ++kt) {
    v16h a = *(const v16h*)(aBase + kt * 512);
    #pragma unroll
    for (int nt = 0; nt < NT1; ++nt) {
      v16h b = *(const v16h*)(w1l + (kt * NT1 + nt) * 512 + lane * 16);
      acc[nt] = __builtin_amdgcn_wmma_f32_16x16x32_f16(
          false, a, false, b, (short)0, acc[nt], false, false);
    }
  }

  _Float16* h = hbuf + wave * 16 * FC;
  #pragma unroll
  for (int nt = 0; nt < NT1; ++nt) {
    float bias = b1[nt * 16 + nlo];
    #pragma unroll
    for (int r = 0; r < 8; ++r) {
      float v = acc[nt][r] + bias;
      h[(r + mhi) * FC + nt * 16 + nlo] = (_Float16)(v > 0.0f ? v : 0.0f);
    }
  }
  __builtin_amdgcn_wave_barrier();   // DS ops are wave-in-order; pin compiler order

  // ---------------- layer 2: [16x128] x [128x128] ----------------
  v8f acc2[NT1];
  #pragma unroll
  for (int nt = 0; nt < NT1; ++nt) acc2[nt] = zero;
  #pragma unroll 1                         // rolled: keep register pressure low
  for (int kt = 0; kt < KT2; ++kt) {
    union { v16h v; unsigned int u[8]; } a2;
    #pragma unroll
    for (int vg = 0; vg < 8; ++vg) {
      int j  = vg * 2;
      int k0 = (j & 7) | ((j & 8) << 1) | khi;     // even K of the half-pair
      a2.u[vg] = *(const unsigned int*)(h + mA * FC + kt * 32 + k0);
    }
    #pragma unroll
    for (int nt = 0; nt < NT1; ++nt) {
      v16h b = *(const v16h*)(w2l + (kt * NT1 + nt) * 512 + lane * 16);
      acc2[nt] = __builtin_amdgcn_wmma_f32_16x16x32_f16(
          false, a2.v, false, b, (short)0, acc2[nt], false, false);
    }
  }
  __builtin_amdgcn_wave_barrier();
  #pragma unroll
  for (int nt = 0; nt < NT1; ++nt) {
    float bias = b2[nt * 16 + nlo];
    #pragma unroll
    for (int r = 0; r < 8; ++r) {
      float v = acc2[nt][r] + bias;
      h[(r + mhi) * FC + nt * 16 + nlo] = (_Float16)(v > 0.0f ? v : 0.0f);
    }
  }
  __builtin_amdgcn_wave_barrier();

  // ---------------- layer 3: [16x128] x [128x16] (3 valid cols) ----------------
  v8f acc3 = zero;
  #pragma unroll
  for (int kt = 0; kt < KT2; ++kt) {
    union { v16h v; unsigned int u[8]; } a3;
    #pragma unroll
    for (int vg = 0; vg < 8; ++vg) {
      int j  = vg * 2;
      int k0 = (j & 7) | ((j & 8) << 1) | khi;
      a3.u[vg] = *(const unsigned int*)(h + mA * FC + kt * 32 + k0);
    }
    v16h b = *(const v16h*)(w3l + kt * 512 + lane * 16);
    acc3 = __builtin_amdgcn_wmma_f32_16x16x32_f16(
        false, a3.v, false, b, (short)0, acc3, false, false);
  }

  // sigmoid, weight by compositing weights, reduce deterministically
  float bias3   = (nlo < 3) ? b3[nlo] : 0.0f;
  float partial = 0.0f;
  #pragma unroll
  for (int r = 0; r < 8; ++r) {
    float v = acc3[r] + bias3;
    float c = 1.0f / (1.0f + __expf(-v));
    int   s = wave * 16 + r + mhi;               // sample index within ray
    partial += wray[s] * c;
  }
  partial += __shfl_down(partial, 16);           // fold rows 8..15 into lanes 0..15
  if (lane < 3) wsum[wave * 3 + lane] = partial;
  __syncthreads();
  if (tid < 3) {
    float acc_rgb = 0.0f;
    #pragma unroll
    for (int w = 0; w < 8; ++w) acc_rgb += wsum[w * 3 + tid];
    out[ray * 3 + tid] = acc_rgb;
  }
}

// ---------------------------------------------------------------------------
extern "C" void kernel_launch(void* const* d_in, const int* in_sizes, int n_in,
                              void* d_out, int out_size, void* d_ws, size_t ws_size,
                              hipStream_t stream) {
  const float* xyz          = (const float*)d_in[0];
  const float* viewdirs     = (const float*)d_in[1];
  const float* dists        = (const float*)d_in[2];
  const float* density_grid = (const float*)d_in[3];
  const float* app_grid     = (const float*)d_in[4];
  const float* w1 = (const float*)d_in[5];
  const float* b1 = (const float*)d_in[6];
  const float* w2 = (const float*)d_in[7];
  const float* b2 = (const float*)d_in[8];
  const float* w3 = (const float*)d_in[9];
  const float* b3 = (const float*)d_in[10];
  float* out = (float*)d_out;

  char* ws = (char*)d_ws;
  _Float16* wall  = (_Float16*)(ws + WS_WALL);
  float*    sigma = (float*)   (ws + WS_SIG);
  _Float16* afrag = (_Float16*)(ws + WS_AFRAG);

  prep_weights<<<(WALL_H + 255) / 256, 256, 0, stream>>>(w1, w2, w3, wall);

  for (int ch = 0; ch < NCHUNK; ++ch) {
    int ray0 = ch * CH_RAYS;
    featurize<<<CH_PTS / 256, 256, 0, stream>>>(
        xyz, viewdirs, density_grid, app_grid, afrag, sigma, ray0);
    mlp_render<<<CH_RAYS, 256, 0, stream>>>(
        afrag, wall, b1, b2, b3, sigma, dists, out, ray0);
  }
}
